// PPRGNN_PPI_38981123178698
// MI455X (gfx1250) — matrix operand order repaired and checked by reading, not compiled, not run on hardware
//
#include <hip/hip_runtime.h>
#include <cstddef>

// ---------------------------------------------------------------------------
// PPRGNN (PPI): 5 x [ H = X W^T + b ; 10x Z <- 0.9*A Z + 0.1*H ;
//                     y = Z + X Vw^T + Vb ; ELU (except last) ]
// N = 50000 (== 3125 * 16 exactly), E = 1.6M
// GEMMs run on V_WMMA_F32_16X16X32_BF16 with zero-padded bf16 operands so the
// inner loop is pure  4x global_load_b128 -> v_wmma  (no branches/converts).
// ---------------------------------------------------------------------------

typedef __attribute__((ext_vector_type(16))) __bf16 v16bf;
typedef __attribute__((ext_vector_type(8)))  __bf16 v8bf;
typedef __attribute__((ext_vector_type(8)))  float  v8f;

#define PPR_N 50000
#define PPR_E 1600000

__device__ __forceinline__ float elu_f(float x) {
    return x > 0.0f ? x : (__expf(x) - 1.0f);
}

// ---------------------------------------------------------------------------
// Zero-padded fp32 -> bf16 repack:  dst[r,c] = (r<srows && c<scols) ? src : 0
// ---------------------------------------------------------------------------
__global__ void __launch_bounds__(256)
ppr_pack_bf16(const float* __restrict__ src, __bf16* __restrict__ dst,
              int srows, int scols, int drows, int dcols)
{
    const int idx = blockIdx.x * 256 + threadIdx.x;
    if (idx >= drows * dcols) return;
    const int r = idx / dcols;
    const int c = idx - r * dcols;
    const float v = (r < srows && c < scols) ? src[(size_t)r * scols + c] : 0.0f;
    dst[idx] = (__bf16)v;
}

// ---------------------------------------------------------------------------
// Dense GEMM via V_WMMA_F32_16X16X32_BF16 on packed bf16 operands.
//   Xb: [N,     dipad] bf16 (row-major, zero-padded K)
//   Wb: [dnpad, dipad] bf16 (row-major, zero-padded K and rows)
// One wave per 16x16 output tile; 8 waves/block cover 128 columns.
// Per-lane WMMA operand layout (contiguous k order, same for A and B):
//   A 16x32: vgpr pairs 0-3 -> k in [half*16, half*16+8)      (one b128 group)
//            vgpr pairs 4-7 -> k in [half*16+8, half*16+16) (one b128 group)
//   B 32x16: lane's 16 k-values = [half*16, half*16+16)      (two b128 groups)
//   C/D:     vgpr r -> row = r + 8*half, col = lane&15
// Epilogue: +bias, +optional addend (in-place-safe), optional ELU,
//           optional fp32 out/out2 stores, optional packed-bf16 store of the
//           activation (becomes next layer's X; pad columns zero-filled).
// ---------------------------------------------------------------------------
__global__ void __launch_bounds__(256)
ppr_gemm_wmma(const __bf16* __restrict__ Xb, const __bf16* __restrict__ Wb,
              const float* __restrict__ bias, const float* __restrict__ addend,
              float* __restrict__ out, float* __restrict__ out2,
              __bf16* __restrict__ outbf,
              int dipad, int dn, int dnpad, int apply_elu)
{
    const int lane = threadIdx.x & 31;
    const int wave = threadIdx.x >> 5;
    const int row_base = blockIdx.x * 16;
    const int col_base = (blockIdx.y * 8 + wave) * 16;
    if (col_base >= dn) return;                  // wave-uniform exit

    const int half = lane >> 4;
    const int mn   = lane & 15;

    const int gpr = dipad >> 3;                  // 16B groups per row
    const v8bf* __restrict__ Xr = (const v8bf*)Xb + (size_t)(row_base + mn) * gpr;
    const v8bf* __restrict__ Wr = (const v8bf*)Wb + (size_t)(col_base + mn) * gpr;

    v8f acc = {0.f, 0.f, 0.f, 0.f, 0.f, 0.f, 0.f, 0.f};

    for (int g = 0; g < gpr; g += 4) {           // 32 K-values per step
        const v8bf alo = Xr[g + 2 * half];
        const v8bf ahi = Xr[g + 2 * half + 1];
        const v8bf blo = Wr[g + 2 * half];
        const v8bf bhi = Wr[g + 2 * half + 1];
        const v16bf a = __builtin_shufflevector(alo, ahi,
                            0,1,2,3,4,5,6,7,8,9,10,11,12,13,14,15);
        const v16bf b = __builtin_shufflevector(blo, bhi,
                            0,1,2,3,4,5,6,7,8,9,10,11,12,13,14,15);
        acc = __builtin_amdgcn_wmma_f32_16x16x32_bf16(
                  false, a, false, b, (short)0, acc, false, false);
    }

    const int col = col_base + mn;
    if (col < dn) {
        const float bv = bias[col];
#pragma unroll
        for (int r = 0; r < 8; ++r) {
            const int row = row_base + r + 8 * half;
            const size_t idx = (size_t)row * dn + col;
            float v = acc[r] + bv;
            if (addend)    v += addend[idx];     // read-before-write: in-place safe
            if (apply_elu) v = elu_f(v);
            if (out)   out[idx]  = v;
            if (out2)  out2[idx] = v;            // dup-store: Z0 = H
            if (outbf) outbf[(size_t)row * dnpad + col] = (__bf16)v;
        }
    } else if (outbf && col < dnpad) {
        // zero-fill bf16 padding columns of the next-layer X
#pragma unroll
        for (int r = 0; r < 8; ++r) {
            const int row = row_base + r + 8 * half;
            outbf[(size_t)row * dnpad + col] = (__bf16)0.0f;
        }
    }
}

// ---------------------------------------------------------------------------
// Z_next = ALPHA * H   (float4; N*dn always divisible by 4)
// ---------------------------------------------------------------------------
__global__ void __launch_bounds__(256)
ppr_scale_init(const float4* __restrict__ H, float4* __restrict__ Zn, int n4)
{
    const int i = blockIdx.x * 256 + threadIdx.x;
    if (i < n4) {
        float4 h = H[i];
        float4 o;
        o.x = 0.1f * h.x; o.y = 0.1f * h.y; o.z = 0.1f * h.z; o.w = 0.1f * h.w;
        Zn[i] = o;
    }
}

// ---------------------------------------------------------------------------
// Z_next[row,:] += 0.9 * w_e * Z[col,:]   — one wave per edge, lanes stride
// the feature dim (coalesced 128B gathers, native f32 atomics into L2).
// ---------------------------------------------------------------------------
__global__ void __launch_bounds__(256)
ppr_spmm_atomic(const int* __restrict__ ei, const float* __restrict__ ew,
                const float* __restrict__ Z, float* __restrict__ Zn, int dn)
{
    const int e = blockIdx.x * 8 + (threadIdx.x >> 5);
    if (e >= PPR_E) return;
    const int lane = threadIdx.x & 31;

    const int row = ei[e];
    const int col = ei[PPR_E + e];
    const float w = 0.9f * ew[e];

    const float* __restrict__ src = Z  + (size_t)col * dn;
    float* __restrict__       dst = Zn + (size_t)row * dn;

    for (int d = lane; d < dn; d += 32) {
        __hip_atomic_fetch_add(&dst[d], w * src[d],
                               __ATOMIC_RELAXED, __HIP_MEMORY_SCOPE_AGENT);
    }
}

// ---------------------------------------------------------------------------
// Host orchestration
// ---------------------------------------------------------------------------
extern "C" void kernel_launch(void* const* d_in, const int* in_sizes, int n_in,
                              void* d_out, int out_size, void* d_ws, size_t ws_size,
                              hipStream_t stream)
{
    (void)in_sizes; (void)n_in; (void)out_size; (void)ws_size;

    const float* features = (const float*)d_in[0];
    const int*   ei       = (const int*)  d_in[1];
    const float* ew       = (const float*)d_in[2];

    const int dis[5] = {50, 512, 256, 256, 128};
    const int dns[5] = {512, 256, 256, 128, 121};

    const float *Wp[5], *bp[5], *Vw[5], *Vb[5];
    for (int i = 0; i < 5; ++i) {
        Wp[i] = (const float*)d_in[3 + 4 * i];
        bp[i] = (const float*)d_in[4 + 4 * i];
        Vw[i] = (const float*)d_in[5 + 4 * i];
        Vb[i] = (const float*)d_in[6 + 4 * i];
    }

    // ---- workspace carve-up (all offsets 256B-aligned) ----
    // 3 fp32 buffers (H, Za, Zb), 2 bf16 X buffers, 2 bf16 weight buffers
    char* ws = (char*)d_ws;
    const size_t fB  = (size_t)PPR_N * 512 * sizeof(float);   // 102.4 MB
    const size_t xB  = (size_t)PPR_N * 512 * sizeof(__bf16);  //  51.2 MB
    const size_t wB  = (size_t)512 * 512 * sizeof(__bf16);    //   0.5 MB
    float*  H     = (float*) (ws);
    float*  Za    = (float*) (ws + fB);
    float*  Zb    = (float*) (ws + 2 * fB);
    __bf16* XbfA  = (__bf16*)(ws + 3 * fB);
    __bf16* XbfB  = (__bf16*)(ws + 3 * fB + xB);
    __bf16* Wbf   = (__bf16*)(ws + 3 * fB + 2 * xB);
    __bf16* Vwbf  = (__bf16*)(ws + 3 * fB + 2 * xB + wB);

    const int row_tiles = PPR_N / 16;            // 3125 exactly

    // pack layer-1 input features -> bf16 [N, 64]
    {
        const int tot = PPR_N * 64;
        ppr_pack_bf16<<<(tot + 255) / 256, 256, 0, stream>>>(
            features, XbfA, PPR_N, 50, PPR_N, 64);
    }

    __bf16* Xc = XbfA;   // current X (bf16 packed)
    __bf16* Xn = XbfB;   // next X

    for (int l = 0; l < 5; ++l) {
        const int di = dis[l], dn = dns[l];
        const int dipad = (di + 31) & ~31;
        const int dnpad = (dn + 31) & ~31;

        // pack both weight matrices for this layer (rows and K zero-padded)
        {
            const int tot = dnpad * dipad;
            ppr_pack_bf16<<<(tot + 255) / 256, 256, 0, stream>>>(
                Wp[l], Wbf, dn, di, dnpad, dipad);
            ppr_pack_bf16<<<(tot + 255) / 256, 256, 0, stream>>>(
                Vw[l], Vwbf, dn, di, dnpad, dipad);
        }

        dim3 gg(row_tiles, (dn + 127) / 128);

        // H = X W^T + b ; dup-store Z0 = H
        ppr_gemm_wmma<<<gg, 256, 0, stream>>>(Xc, Wbf, bp[l], nullptr,
                                              H, Za, nullptr,
                                              dipad, dn, dnpad, 0);

        // 10 fixed-point steps: Z_next = 0.9 * A Z + 0.1 * H
        float* Zc = Za;
        float* Zn = Zb;
        const size_t nel = (size_t)PPR_N * dn;
        const int n4 = (int)(nel / 4);
        const int init_blocks = (n4 + 255) / 256;
        const int spmm_blocks = (PPR_E + 7) / 8;
        for (int it = 0; it < 10; ++it) {
            ppr_scale_init<<<init_blocks, 256, 0, stream>>>(
                (const float4*)H, (float4*)Zn, n4);
            ppr_spmm_atomic<<<spmm_blocks, 256, 0, stream>>>(ei, ew, Zc, Zn, dn);
            float* t = Zc; Zc = Zn; Zn = t;
        }

        // y = Z + X Vw^T + Vb ; ELU except last layer.
        // Inner layers: emit only packed bf16 activation (next X).
        // Last layer:   emit fp32 to d_out.
        float*  yout  = (l == 4) ? (float*)d_out : nullptr;
        __bf16* ybf   = (l < 4) ? Xn : nullptr;
        ppr_gemm_wmma<<<gg, 256, 0, stream>>>(Xc, Vwbf, Vb[l], Zc,
                                              yout, nullptr, ybf,
                                              dipad, dn, dnpad, (l != 4) ? 1 : 0);

        __bf16* t = Xc; Xc = Xn; Xn = t;
    }
}
